// EgoGNN_360777253399
// MI455X (gfx1250) — matrix operand order repaired and checked by reading, not compiled, not run on hardware
//
#include <hip/hip_runtime.h>
#include <hip/hip_bf16.h>

// CDNA5 / gfx1250 EgoGNN pipeline.
// WMMA types (wave32): 16x16 f32 accumulator = v8f, bf16 16x32 A / 32x16 B = v16bf.
typedef __attribute__((ext_vector_type(16))) __bf16 v16bf;
typedef __attribute__((ext_vector_type(8)))  float  v8f;
typedef __attribute__((ext_vector_type(4)))  int    v4i;

#define NFEAT 128   // F_IN == HID == 128 (K of every GEMM)

// gfx1250 async global->LDS copy (ASYNCcnt path), when the toolchain has it.
#if defined(__HIP_DEVICE_COMPILE__) && __has_builtin(__builtin_amdgcn_global_load_async_to_lds_b128)
#define EGO_ASYNC_LDS 1
#else
#define EGO_ASYNC_LDS 0
#endif

// ---------------------------------------------------------------------------
// Zero an f32 buffer (float4 stores, grid-stride).
// ---------------------------------------------------------------------------
__global__ void k_zero(float4* __restrict__ p, long n4) {
  long i = (long)blockIdx.x * blockDim.x + threadIdx.x;
  long stride = (long)gridDim.x * blockDim.x;
  float4 z = {0.f, 0.f, 0.f, 0.f};
  for (; i < n4; i += stride) p[i] = z;
}

// ---------------------------------------------------------------------------
// Edge scatter-add: acc[rows[e]] += x[cols[e]]  (128 feats; 32 lanes x 4 floats
// per edge). x (25.6MB) and acc live in the 192MB L2, so the float4 gather and
// the global_atomic_add_f32 ops stay on-chip.
// ---------------------------------------------------------------------------
__global__ void k_scatter_add(const float* __restrict__ x,
                              const int* __restrict__ rows,
                              const int* __restrict__ cols,
                              float* __restrict__ acc, long nEdges) {
  long t = (long)blockIdx.x * blockDim.x + threadIdx.x;
  long total = nEdges * 32;
  long stride = (long)gridDim.x * blockDim.x;
  for (; t < total; t += stride) {
    long e = t >> 5;
    int lane = (int)(t & 31);
    int r = rows[e];
    int c = cols[e];
    const float4 v = *(const float4*)(x + (long)c * NFEAT + lane * 4);
    float* dst = acc + (long)r * NFEAT + lane * 4;
    atomicAdd(dst + 0, v.x);
    atomicAdd(dst + 1, v.y);
    atomicAdd(dst + 2, v.z);
    atomicAdd(dst + 3, v.w);
  }
}

// ---------------------------------------------------------------------------
// Pre-swizzle a f32 weight (K=128 x outc) into a contiguous-k bf16 B-fragment
// layout for v_wmma_f32_16x16x32_bf16:
//   frag[nt][kk][lane][e]  with  n = nt*16 + (lane&15),
//                               k = kk*32 + (lane>>4)*16 + e.
// Columns beyond outc are zero-padded (classifier 47 -> 48).
// ---------------------------------------------------------------------------
__global__ void k_prep_w(const float* __restrict__ W, int outc,
                         __bf16* __restrict__ frag, int nTiles) {
  int idx = blockIdx.x * blockDim.x + threadIdx.x;
  int total = nTiles * 2048;               // 4 kk * 32 lanes * 16 elems
  if (idx >= total) return;
  int e    = idx & 15;
  int lane = (idx >> 4) & 31;
  int kk   = (idx >> 9) & 3;
  int nt   = idx >> 11;
  int ncol = nt * 16 + (lane & 15);
  int k    = kk * 32 + (lane >> 4) * 16 + e;
  float v = (ncol < outc) ? W[(long)k * outc + ncol] : 0.f;
  frag[idx] = (__bf16)v;
}

__global__ void k_pad_bias(const float* __restrict__ b, float* __restrict__ bp,
                           int outc, int padded) {
  int i = threadIdx.x;
  if (i < padded) bp[i] = (i < outc) ? b[i] : 0.f;
}

// ---------------------------------------------------------------------------
// Fused GEMM: out = act( scaleA*(A [+ A2]) @ W + bias ).
//   256 threads = 8 wave32s; block covers 128 rows, each wave a 16-row tile.
//   K = 128 -> 4 x v_wmma_f32_16x16x32_bf16 per 16-col tile; f32 accum.
//   All row counts are pre-padded to multiples of 128, so there is NO
//   predication anywhere: EXEC stays all-ones (WMMA requirement) and stores
//   are straight-line.
//   Weight fragments staged to LDS once via async global->LDS (ASYNCcnt) when
//   available, else a cooperative uint4 copy; ds_load_b128 feeds the WMMAs.
//   A is converted f32->bf16 on the fly, fusing the 1/N ego scale or the
//   GIN (x + aggr) elementwise add into the conversion.
// A-fragment layout: lane M = lane&15, h = lane>>4; elems 0..15 hold
// K = kk*32 + 16h + e (matches the B-fragment packing).
// C/D ISA layout: reg r, row = r + 8h, col = lane&15.
// ---------------------------------------------------------------------------
template <int NT, bool HAS_A2, bool RELU>
__global__ __launch_bounds__(256)
void k_gemm_bias_act(const float* __restrict__ A, const float* __restrict__ A2,
                     float scaleA,
                     const __bf16* __restrict__ fragW,
                     const float* __restrict__ bias,
                     float* __restrict__ out, int outLd) {
  __shared__ __attribute__((aligned(64))) __bf16 smem[NT * 4 * 32 * 16];
  constexpr int kWVec = NT * 4 * 32 * 16 * 2 / 16;   // 16B chunks (<= 2048)

  int tid = threadIdx.x;
#if EGO_ASYNC_LDS
  {
    // Builtin signature (per compiler diagnostic): AS1 int4* src, AS3 int4* dst,
    // imm offset, imm cpol. One b128 transfer per lane per call.
    auto gsrc = (__attribute__((address_space(1))) v4i*)fragW;
    auto ldst = (__attribute__((address_space(3))) v4i*)smem;
    #pragma unroll
    for (int i = 0; i < kWVec / 256; ++i) {
      int idx = i * 256 + tid;
      __builtin_amdgcn_global_load_async_to_lds_b128(gsrc + idx, ldst + idx, 0, 0);
    }
#if __has_builtin(__builtin_amdgcn_s_wait_asynccnt)
    __builtin_amdgcn_s_wait_asynccnt(0);
#else
    asm volatile("s_wait_asynccnt 0" ::: "memory");
#endif
  }
#else
  {
    const uint4* wsrc = (const uint4*)fragW;
    uint4* wdst = (uint4*)smem;
    #pragma unroll
    for (int i = 0; i < kWVec / 256; ++i)
      wdst[i * 256 + tid] = wsrc[i * 256 + tid];
  }
#endif
  __syncthreads();

  int wave = tid >> 5;
  int lane = tid & 31;
  int m = lane & 15;
  int h = lane >> 4;
  long tileRow0 = (long)blockIdx.x * 128 + wave * 16;
  const float* arow  = A + (tileRow0 + m) * NFEAT;
  const float* arow2 = HAS_A2 ? (A2 + (tileRow0 + m) * NFEAT) : nullptr;
  __builtin_prefetch(arow, 0, 0);                      // global_prefetch_b8

  // Build the four A fragments (one per 32-wide K step).
  v16bf afrag[4];
#pragma unroll
  for (int kk = 0; kk < 4; ++kk) {
    int b1 = kk * 32 + 16 * h;
    int b2 = b1 + 8;
    float tmp[16];
    *(float4*)(tmp + 0)  = *(const float4*)(arow + b1);
    *(float4*)(tmp + 4)  = *(const float4*)(arow + b1 + 4);
    *(float4*)(tmp + 8)  = *(const float4*)(arow + b2);
    *(float4*)(tmp + 12) = *(const float4*)(arow + b2 + 4);
    if constexpr (HAS_A2) {                            // fused GIN x + aggr
      float t2[16];
      *(float4*)(t2 + 0)  = *(const float4*)(arow2 + b1);
      *(float4*)(t2 + 4)  = *(const float4*)(arow2 + b1 + 4);
      *(float4*)(t2 + 8)  = *(const float4*)(arow2 + b2);
      *(float4*)(t2 + 12) = *(const float4*)(arow2 + b2 + 4);
#pragma unroll
      for (int e = 0; e < 16; ++e) tmp[e] += t2[e];
    }
#pragma unroll
    for (int e = 0; e < 16; ++e) afrag[kk][e] = (__bf16)(tmp[e] * scaleA);
  }

  const v16bf* wfrag = (const v16bf*)smem;
#pragma unroll
  for (int nt = 0; nt < NT; ++nt) {
    float bval = bias[nt * 16 + m];
    v8f acc = {};
#pragma unroll
    for (int r = 0; r < 8; ++r) acc[r] = bval;          // bias pre-loaded into C
#pragma unroll
    for (int kk = 0; kk < 4; ++kk) {
      v16bf bfrag = wfrag[(nt * 4 + kk) * 32 + lane];   // ds_load_b128 x2
      acc = __builtin_amdgcn_wmma_f32_16x16x32_bf16(
          /*neg_a=*/false, afrag[kk], /*neg_b=*/false, bfrag,
          /*c_mod=*/(short)0, acc, /*reuse_a=*/false, /*reuse_b=*/false);
    }
    int col = nt * 16 + m;
    float* obase = out + (tileRow0 + 8 * h) * (long)outLd + col;
#pragma unroll
    for (int r = 0; r < 8; ++r) {
      float v = acc[r];
      if constexpr (RELU) v = fmaxf(v, 0.f);
      obase[(long)r * outLd] = v;                       // unguarded: rows padded
    }
  }
}

// ---------------------------------------------------------------------------
// Row-wise log_softmax over the first C of ld columns.
// ---------------------------------------------------------------------------
__global__ void k_logsoftmax(const float* __restrict__ logits,
                             float* __restrict__ out, int n, int C, int ld) {
  int i = blockIdx.x * blockDim.x + threadIdx.x;
  if (i >= n) return;
  const float* r = logits + (long)i * ld;
  float mx = -3.402823466e38f;
  for (int j = 0; j < C; ++j) mx = fmaxf(mx, r[j]);
  float s = 0.f;
  for (int j = 0; j < C; ++j) s += __expf(r[j] - mx);
  float ls = __logf(s);
  for (int j = 0; j < C; ++j) out[(long)i * C + j] = r[j] - mx - ls;
}

// ---------------------------------------------------------------------------
// Launch: 4 weight preps + padded bias, then per layer: zero / scatter / GEMM,
// then log_softmax. All on `stream`, no allocs, graph-capture safe.
// Row count is padded to a multiple of 128 inside d_ws so every GEMM block is
// full (pad rows of the accumulator buffer are zeroed -> pad outputs are just
// bias values, never read by the final log_softmax over n rows).
// ---------------------------------------------------------------------------
extern "C" void kernel_launch(void* const* d_in, const int* in_sizes, int n_in,
                              void* d_out, int out_size, void* d_ws, size_t ws_size,
                              hipStream_t stream) {
  const float* x_in   = (const float*)d_in[0];
  const int*   ei     = (const int*)d_in[1];   // [2][E]     (src, tgt)
  const int*   ego    = (const int*)d_in[2];   // [2][EGO_E] (row, col)
  const float* W_ego0 = (const float*)d_in[3];
  const float* b_ego0 = (const float*)d_in[4];
  const float* W_gin0 = (const float*)d_in[5];
  const float* b_gin0 = (const float*)d_in[6];
  const float* W_ego1 = (const float*)d_in[7];
  const float* b_ego1 = (const float*)d_in[8];
  const float* W_gin1 = (const float*)d_in[9];
  const float* b_gin1 = (const float*)d_in[10];

  int  n    = in_sizes[0] / NFEAT;       // 50000
  int  nPad = (n + 127) & ~127;          // 50048: full 128-row GEMM blocks
  long E_   = in_sizes[1] / 2;           // 600000
  long EGO  = in_sizes[2] / 2;           // 1200000
  int  OUTC = in_sizes[10];              // 47
  int  OUTP = (OUTC + 15) & ~15;         // 48 (3 WMMA column tiles)

  // Workspace carve-up (16B-aligned): 3 x nPad*128 f32 ping-pong buffers,
  // then bf16 weight fragments + padded classifier bias.
  char* ws = (char*)d_ws;
  size_t featBytes = (size_t)nPad * NFEAT * sizeof(float);
  float* bufA = (float*)(ws);
  float* bufB = (float*)(ws + featBytes);
  float* bufC = (float*)(ws + 2 * featBytes);
  char* wp = ws + 3 * featBytes;
  __bf16* fw0 = (__bf16*)wp; wp += 8 * 2048 * sizeof(__bf16);
  __bf16* fw1 = (__bf16*)wp; wp += 8 * 2048 * sizeof(__bf16);
  __bf16* fw2 = (__bf16*)wp; wp += 8 * 2048 * sizeof(__bf16);
  __bf16* fw3 = (__bf16*)wp; wp += 8 * 2048 * sizeof(__bf16);
  float* b3   = (float*)wp;

  dim3 blk(256);
  long n4      = (long)nPad * NFEAT / 4;
  int gZero    = (int)((n4 + 255) / 256);
  int gScatE   = (int)((E_ * 32 + 255) / 256);
  int gScatEgo = (int)((EGO * 32 + 255) / 256);
  int gGemm    = nPad / 128;
  int gPrep8   = (8 * 2048 + 255) / 256;
  int gPrep3   = (3 * 2048 + 255) / 256;
  int gSm      = (n + 255) / 256;
  float invN   = 1.0f / (float)n;

  // Weight fragment prep (bf16 swizzle) + padded classifier bias.
  k_prep_w<<<gPrep8, blk, 0, stream>>>(W_ego0, NFEAT, fw0, 8);
  k_prep_w<<<gPrep8, blk, 0, stream>>>(W_gin0, NFEAT, fw1, 8);
  k_prep_w<<<gPrep8, blk, 0, stream>>>(W_ego1, NFEAT, fw2, 8);
  k_prep_w<<<gPrep3, blk, 0, stream>>>(W_gin1, OUTC,  fw3, 3);
  k_pad_bias<<<1, 64, 0, stream>>>(b_gin1, b3, OUTC, OUTP);

  // Layer 0: Ego conv -> relu(Linear)   [scale 1/N fused into bf16 convert]
  k_zero<<<gZero, blk, 0, stream>>>((float4*)bufA, n4);
  k_scatter_add<<<gScatEgo, blk, 0, stream>>>(x_in, ego, ego + EGO, bufA, EGO);
  k_gemm_bias_act<8, false, true><<<gGemm, blk, 0, stream>>>(
      bufA, nullptr, invN, fw0, b_ego0, bufB, NFEAT);
  // Layer 0: GIN -> relu((x + aggr) @ W)
  k_zero<<<gZero, blk, 0, stream>>>((float4*)bufA, n4);
  k_scatter_add<<<gScatE, blk, 0, stream>>>(bufB, ei + E_, ei, bufA, E_);
  k_gemm_bias_act<8, true, true><<<gGemm, blk, 0, stream>>>(
      bufB, bufA, 1.0f, fw1, b_gin0, bufC, NFEAT);
  // Layer 1: Ego conv -> relu(Linear)
  k_zero<<<gZero, blk, 0, stream>>>((float4*)bufA, n4);
  k_scatter_add<<<gScatEgo, blk, 0, stream>>>(bufC, ego, ego + EGO, bufA, EGO);
  k_gemm_bias_act<8, false, true><<<gGemm, blk, 0, stream>>>(
      bufA, nullptr, invN, fw2, b_ego1, bufB, NFEAT);
  // Layer 1: GIN classifier (no relu), padded to 48 cols.
  k_zero<<<gZero, blk, 0, stream>>>((float4*)bufA, n4);
  k_scatter_add<<<gScatE, blk, 0, stream>>>(bufB, ei + E_, ei, bufA, E_);
  k_gemm_bias_act<3, true, false><<<gGemm, blk, 0, stream>>>(
      bufB, bufA, 1.0f, fw3, b3, bufC, OUTP);
  // log_softmax over 47 classes -> d_out (N x 47 f32)
  k_logsoftmax<<<gSm, blk, 0, stream>>>(bufC, (float*)d_out, n, OUTC, OUTP);
}
